// AttnBlock_1692217114781
// MI455X (gfx1250) — matrix operand order repaired and checked by reading, not compiled, not run on hardware
//
#include <hip/hip_runtime.h>
#include <cstdint>
#include <cstddef>

// ---------------------------------------------------------------------------
// AttnBlock for MI455X (gfx1250): GroupNorm -> QKV GEMMs -> flash attention ->
// proj + residual.  All heavy math via v_wmma_f32_16x16x32_bf16 (wave32).
// B=2, C=512, H=W=64, N=4096, G=32.
//
// Layout strategy: activations stored token-major ([N][C]) so every WMMA
// fragment load is contiguous along the contraction dim (2 x b128 per frag):
//   - GEMM:   D = A[wout][c] x hT[n][c]   (both operands contiguous in c)
//   - scores: S = qT[i][c]  x kT[j][c]    (both contiguous in c)
//   - PV:     O = P[i][j]   x v[c][j]     (v kept channel-major: contiguous j)
// GEMM tiles are DMA'd into double-buffered LDS by the Tensor Data Mover
// (tensor_load_to_lds, TENSORcnt-synchronized, one k-step ahead of compute).
// Workspace: 4 bf16 weights (2MB) + hT/oT (8MB) + qT,kT,v (24MB) = 34MB.
// ---------------------------------------------------------------------------

typedef __attribute__((ext_vector_type(16))) __bf16 v16bf;
typedef __attribute__((ext_vector_type(8)))  __bf16 v8bf;
typedef __attribute__((ext_vector_type(8)))  float  v8f;
typedef __attribute__((ext_vector_type(4)))  float  v4f;
typedef unsigned int v4u __attribute__((ext_vector_type(4)));
typedef int          v8i __attribute__((ext_vector_type(8)));
typedef int          v4i __attribute__((ext_vector_type(4)));

constexpr int BATCH  = 2;
constexpr int C      = 512;
constexpr int N      = 4096;
constexpr int GROUPS = 32;
constexpr int CPG    = C / GROUPS;          // 16 channels per group

#if __has_builtin(__builtin_amdgcn_tensor_load_to_lds) && \
    __has_builtin(__builtin_amdgcn_s_wait_tensorcnt)
#define USE_TDM 1
#else
#define USE_TDM 0
#endif

static __device__ __forceinline__ v16bf cat8(v8bf lo, v8bf hi) {
  return __builtin_shufflevector(lo, hi, 0,1,2,3,4,5,6,7,8,9,10,11,12,13,14,15);
}

static __device__ __forceinline__ v16bf frag16(const __bf16* p) {
  return cat8(*(const v8bf*)p, *(const v8bf*)(p + 8));
}

static __device__ __forceinline__ v8f wmma_bf16(v16bf a, v16bf b, v8f c) {
  // D(16x16 f32) = A(16x32 bf16) x B(32x16 bf16) + C
  return __builtin_amdgcn_wmma_f32_16x16x32_bf16(
      false, a, false, b, (short)0, c, false, false);
}

#if USE_TDM
// TDM: DMA a [64 rows x 32 cols] bf16 tile (row stride = 512 elements) from
// `gsrc` into LDS at `lds_byte_addr`.  D# bit layout per CDNA5 ISA §8.3/8.4:
//  g0: count=1 | lds_addr | global_addr[56:0] | type=2
//  g1: data_size=2B, tensor_dim0=512, tensor_dim1=dim1, tile=32x64, stride=512
static __device__ __forceinline__ void tdm_load_tile_64x32(
    const __bf16* gsrc, unsigned int lds_byte_addr, unsigned int dim1) {
  const unsigned long long ga = (unsigned long long)(uintptr_t)gsrc;
  v4u g0;
  g0[0] = 1u;                                     // count=1, user descriptor
  g0[1] = lds_byte_addr;                          // lds_addr
  g0[2] = (unsigned int)(ga & 0xFFFFFFFFu);       // global_addr[31:0]
  g0[3] = (unsigned int)((ga >> 32) & 0x01FFFFFFu) | (2u << 30); // type=2
  v8i g1;
  g1[0] = (int)(1u << 16);       // workgroup_mask=0, data_size=1 (2 bytes)
  g1[1] = (int)(512u << 16);     // tensor_dim0[15:0]=512  (bits 63:48)
  g1[2] = (int)(dim1 << 16);     // tensor_dim0 hi=0 ; tensor_dim1[15:0]
  g1[3] = (int)(32u << 16);      // tensor_dim1 hi=0 ; tile_dim0=32
  g1[4] = 64;                    // tile_dim1=64, tile_dim2=0
  g1[5] = 512;                   // tensor_dim0_stride lo = 512
  g1[6] = 0;
  g1[7] = 0;
  v4i z4 = {0, 0, 0, 0};
#if defined(__clang_major__) && (__clang_major__ >= 23)
  v8i z8 = {0, 0, 0, 0, 0, 0, 0, 0};
  __builtin_amdgcn_tensor_load_to_lds(g0, g1, z4, z4, z8, 0);
#else
  __builtin_amdgcn_tensor_load_to_lds(g0, g1, z4, z4, 0);
#endif
}
#endif

// ---------------------------------------------------------------------------
// Kernel 0: fp32 -> bf16 conversion of the four C x C weight matrices
// ---------------------------------------------------------------------------
__global__ void convert_weights_kernel(const float* __restrict__ wq,
                                       const float* __restrict__ wk,
                                       const float* __restrict__ wv,
                                       const float* __restrict__ wp,
                                       __bf16* __restrict__ dst) {
  const int i = blockIdx.x * 256 + threadIdx.x;
  if (i < C * C) {
    dst[0 * C * C + i] = (__bf16)wq[i];
    dst[1 * C * C + i] = (__bf16)wk[i];
    dst[2 * C * C + i] = (__bf16)wv[i];
    dst[3 * C * C + i] = (__bf16)wp[i];
  }
}

// ---------------------------------------------------------------------------
// Kernel 1: GroupNorm(32, C, eps=1e-6) + affine; output TOKEN-MAJOR bf16
// hT[b][n][c].  One block per (batch, group); transpose via 16x16 LDS tile
// so both global reads and writes stay coalesced.
// ---------------------------------------------------------------------------
__global__ __launch_bounds__(256)
void groupnorm_kernel(const float* __restrict__ x,
                      const float* __restrict__ gsc,
                      const float* __restrict__ gbs,
                      __bf16* __restrict__ hT) {
  const int tid = threadIdx.x;
  const int b   = blockIdx.x / GROUPS;
  const int g   = blockIdx.x % GROUPS;
  const int gc0 = g * CPG;
  const float* xg = x + (size_t)(b * C + gc0) * N;   // 16 x 4096 contiguous

  __shared__ float sS[256];
  __shared__ float sQ[256];
  __shared__ float s_mean, s_rstd;
  __shared__ __bf16 Tt[16][17];                      // transpose tile (padded)

  float s = 0.f, q = 0.f;
#pragma unroll 4
  for (int it = 0; it < 64; ++it) {                  // 64 x 256 float4
    const int idx4 = tid + it * 256;
    v4f v = *(const v4f*)(xg + (size_t)idx4 * 4);
    s += v[0] + v[1] + v[2] + v[3];
    q += v[0]*v[0] + v[1]*v[1] + v[2]*v[2] + v[3]*v[3];
  }
  sS[tid] = s; sQ[tid] = q;
  __syncthreads();
  for (int o = 128; o > 0; o >>= 1) {
    if (tid < o) { sS[tid] += sS[tid + o]; sQ[tid] += sQ[tid + o]; }
    __syncthreads();
  }
  if (tid == 0) {
    const float inv  = 1.0f / (float)(CPG * N);
    const float mean = sS[0] * inv;
    const float var  = sQ[0] * inv - mean * mean;
    s_mean = mean;
    s_rstd = rsqrtf(var + 1e-6f);
  }
  __syncthreads();
  const float mean = s_mean, rstd = s_rstd;

  const int cl = tid >> 4, nl = tid & 15;            // read mapping
  const float sc = gsc[gc0 + cl] * rstd;
  const float bs = gbs[gc0 + cl] - mean * sc;        // (v-mean)*rstd*g + b

  for (int win = 0; win < N / 16; ++win) {
    const float v = xg[(size_t)cl * N + win * 16 + nl];
    Tt[nl][cl] = (__bf16)(v * sc + bs);
    __syncthreads();
    hT[((size_t)b * N + win * 16 + (tid >> 4)) * C + gc0 + (tid & 15)] =
        Tt[tid >> 4][tid & 15];
    __syncthreads();
  }
}

// ---------------------------------------------------------------------------
// Kernel 2: GEMM between weight A[Cout][Cin] and token-major act BT[N][C].
// Block tile 64x64, 4 waves.  Both 64x32 LDS tiles are DMA'd by the TDM into
// double-buffered LDS one k-step ahead of compute (TENSORcnt synchronized);
// the inner loop is pure ds_load_b128 + v_wmma.
//  mode 0: D[m][n]   -> bf16 out[C][N]              (v)
//  mode 1: D[m][n]   -> f32  out[C][N] + x residual (projection)
//  mode 2: D[n][m]   -> bf16 out[N][C] token-major  (q, k)
// ---------------------------------------------------------------------------
__global__ __launch_bounds__(128)
void gemm_wmma_kernel(const __bf16* __restrict__ A,
                      const __bf16* __restrict__ BT,
                      const float*  __restrict__ bias,
                      const float*  __restrict__ xres,
                      float*        __restrict__ outf,
                      __bf16*       __restrict__ outbf,
                      float alpha, int mode) {
  const int tid  = threadIdx.x;
  const int lane = tid & 31;
  const int w    = tid >> 5;
  const int hi   = lane >> 4;
  const int lo   = lane & 15;
  const int n0   = blockIdx.x * 64;        // token tile
  const int m0   = blockIdx.y * 64;        // output-channel tile
  const int bz   = blockIdx.z;
  const __bf16* Bb = BT + (size_t)bz * N * C;

  __shared__ __align__(16) __bf16 Ws[2][64 * 32];   // weight tiles [m][c]
  __shared__ __align__(16) __bf16 Ts[2][64 * 32];   // token tiles  [n][c]

  v8f acc[4];
  const v8f zero8 = {0.f,0.f,0.f,0.f,0.f,0.f,0.f,0.f};
#pragma unroll
  for (int s = 0; s < 4; ++s) acc[s] = zero8;

#if USE_TDM
  if (w == 0) {    // prologue: DMA first k-step tiles into buffer 0
    tdm_load_tile_64x32(A  + (size_t)m0 * C, (unsigned)(uintptr_t)&Ws[0][0], 512u);
    tdm_load_tile_64x32(Bb + (size_t)n0 * C, (unsigned)(uintptr_t)&Ts[0][0],
                        (unsigned)N);
  }
#endif

  for (int kk = 0; kk < C; kk += 32) {
    const int p = (kk >> 5) & 1;
#if USE_TDM
    if (w == 0) __builtin_amdgcn_s_wait_tensorcnt(0);
    __syncthreads();
    if (w == 0 && kk + 32 < C) {     // DMA next k-step into other buffer
      tdm_load_tile_64x32(A  + (size_t)m0 * C + kk + 32,
                          (unsigned)(uintptr_t)&Ws[p ^ 1][0], 512u);
      tdm_load_tile_64x32(Bb + (size_t)n0 * C + kk + 32,
                          (unsigned)(uintptr_t)&Ts[p ^ 1][0], (unsigned)N);
    }
#else
    {   // fallback: per-thread 128-bit staging (straight copies)
      const int row = tid >> 1, cb = (tid & 1) * 16;
      *(v8bf*)&Ws[p][row * 32 + cb] =
          *(const v8bf*)(A + (size_t)(m0 + row) * C + kk + cb);
      *(v8bf*)&Ts[p][row * 32 + cb] =
          *(const v8bf*)(Bb + (size_t)(n0 + row) * C + kk + cb);
    }
    __syncthreads();
#endif

    // fragment roles swap between modes (D rows = channels vs tokens)
    const __bf16* a_tile = (mode == 2) ? Ts[p] : Ws[p];
    const __bf16* b_tile = (mode == 2) ? Ws[p] : Ts[p];

    // A fragment: lane = row, halves K = {a0..a0+15}
    const int arow = w * 16 + lo;
    const int a0   = hi * 16;
    v16bf af = cat8(*(const v8bf*)&a_tile[arow * 32 + a0],
                    *(const v8bf*)&a_tile[arow * 32 + a0 + 8]);
#pragma unroll
    for (int s = 0; s < 4; ++s) {
      // B fragment: lane = col, halves along K with kb = hi*16
      const int col = s * 16 + lo;
      const int kb  = hi * 16;
      acc[s] = wmma_bf16(af, frag16(&b_tile[col * 32 + kb]), acc[s]);
    }
    __syncthreads();
  }

  // ---- epilogue (D layout: row = r + 8*hi, col = lane&15) ----
#pragma unroll
  for (int s = 0; s < 4; ++s) {
#pragma unroll
    for (int r = 0; r < 8; ++r) {
      const int row = w * 16 + r + 8 * hi;
      if (mode == 2) {
        const int token = n0 + row;
        const int ch    = m0 + s * 16 + lo;
        const size_t idx = ((size_t)bz * N + token) * C + ch;
        outbf[idx] = (__bf16)(alpha * (acc[s][r] + bias[ch]));
      } else {
        const int m = m0 + row;
        const int n = n0 + s * 16 + lo;
        const size_t idx = ((size_t)bz * C + m) * N + n;
        const float val = acc[s][r] + bias[m];
        if (mode == 0) outbf[idx] = (__bf16)(alpha * val);
        else           outf[idx]  = xres[idx] + val;
      }
    }
  }
}

// ---------------------------------------------------------------------------
// Kernel 3: flash attention.  Block = 16 query rows, 128 threads (4 waves);
// wave w owns a 128-wide chunk of the C=512 head dim.  KV streamed in
// 32-column chunks; per chunk: partial scores (8 wmma) reduced across waves
// in LDS, online softmax, P re-read as A-frags, P*V^T (8 wmma).
// Q/K/V fragments load straight from global (qkv = 24MB, L2 resident).
// ---------------------------------------------------------------------------
__global__ __launch_bounds__(128)
void attn_flash_kernel(const __bf16* __restrict__ qT,   // [B][N][C]
                       const __bf16* __restrict__ kT,   // [B][N][C]
                       const __bf16* __restrict__ vm,   // [B][C][N]
                       __bf16*       __restrict__ oT) { // [B][N][C]
  const int tid  = threadIdx.x;
  const int lane = tid & 31;
  const int w    = tid >> 5;
  const int hi   = lane >> 4;
  const int lo   = lane & 15;
  const int qbase = blockIdx.x * 16;
  const int b     = blockIdx.y;
  const size_t tok_off = (size_t)b * N * C;
  const __bf16* Q = qT + tok_off;
  const __bf16* K = kT + tok_off;
  const __bf16* V = vm + tok_off;             // same element count per batch

  __shared__ float  Sred[4][16][32];          // per-wave partial scores
  __shared__ float  Sfull[16][32];
  __shared__ __align__(16) __bf16 Pbuf[16][32];
  __shared__ float  row_m[16], row_l[16], row_sf[16];

  if (tid < 16) { row_m[tid] = -1e30f; row_l[tid] = 0.f; }

  const int c0 = w * 128;                     // this wave's C-chunk
  v16bf qf[4];                                // persistent Q fragments
#pragma unroll
  for (int kc = 0; kc < 4; ++kc) {
    const __bf16* qp = Q + (size_t)(qbase + lo) * C + c0 + kc * 32 + hi * 16;
    qf[kc] = cat8(*(const v8bf*)qp, *(const v8bf*)(qp + 8));
  }

  v8f oacc[8];
  const v8f zero8 = {0.f,0.f,0.f,0.f,0.f,0.f,0.f,0.f};
#pragma unroll
  for (int s = 0; s < 8; ++s) oacc[s] = zero8;
  __syncthreads();

  for (int jb = 0; jb < N; jb += 32) {
    // prefetch next K chunk (this wave's C-range) while computing this one
    if (jb + 32 < N)
      __builtin_prefetch(K + (size_t)(jb + 32 + lo) * C + c0, 0, 0);

    // ---- partial scores over this wave's 128-wide C-range ----
    v8f sacc[2] = {zero8, zero8};
#pragma unroll
    for (int kc = 0; kc < 4; ++kc) {
#pragma unroll
      for (int s = 0; s < 2; ++s) {
        const int col = s * 16 + lo;          // key index within chunk
        // B-frag: halves along contraction c, contiguous in kT row
        const __bf16* kp = K + (size_t)(jb + col) * C + c0 + kc * 32 + hi * 16;
        sacc[s] = wmma_bf16(qf[kc], frag16(kp), sacc[s]);
      }
    }
#pragma unroll
    for (int s = 0; s < 2; ++s)
#pragma unroll
      for (int r = 0; r < 8; ++r)
        Sred[w][r + 8 * hi][s * 16 + lo] = sacc[s][r];
    __syncthreads();

    // ---- cross-wave reduction to full 16x32 score tile ----
#pragma unroll
    for (int u = 0; u < 4; ++u) {
      const int idx = tid + u * 128;
      const int ri = idx >> 5, cj = idx & 31;
      Sfull[ri][cj] = Sred[0][ri][cj] + Sred[1][ri][cj] +
                      Sred[2][ri][cj] + Sred[3][ri][cj];
    }
    __syncthreads();

    // ---- online softmax (one thread per query row) ----
    if (tid < 16) {
      const float mo = row_m[tid];
      float mx = mo;
      for (int j = 0; j < 32; ++j) mx = fmaxf(mx, Sfull[tid][j]);
      const float sf = __expf(mo - mx);
      float l = row_l[tid] * sf;
      for (int j = 0; j < 32; ++j) {
        const float p = __expf(Sfull[tid][j] - mx);
        l += p;
        Pbuf[tid][j] = (__bf16)p;
      }
      row_m[tid] = mx; row_l[tid] = l; row_sf[tid] = sf;
    }
    __syncthreads();

    // ---- rescale O and accumulate P * V^T over this wave's C-chunk ----
    {
      const __bf16* pp = &Pbuf[lo][hi * 16];
      v16bf pf = cat8(*(const v8bf*)pp, *(const v8bf*)(pp + 8));
#pragma unroll
      for (int s = 0; s < 8; ++s) {
#pragma unroll
        for (int r = 0; r < 8; ++r) oacc[s][r] *= row_sf[r + 8 * hi];
        const int cch = c0 + s * 16 + lo;
        // B-frag: halves along contraction j, contiguous in v row
        const __bf16* vp = V + (size_t)cch * N + jb + hi * 16;
        oacc[s] = wmma_bf16(pf, frag16(vp), oacc[s]);
      }
    }
    __syncthreads();
  }

  // ---- finalize: divide by l, store token-major (coalesced in c) ----
#pragma unroll
  for (int s = 0; s < 8; ++s) {
#pragma unroll
    for (int r = 0; r < 8; ++r) {
      const int i = r + 8 * hi;
      const size_t idx = tok_off + (size_t)(qbase + i) * C + c0 + s * 16 + lo;
      oT[idx] = (__bf16)(oacc[s][r] / row_l[i]);
    }
  }
}

// ---------------------------------------------------------------------------
// Host-side launch
// ---------------------------------------------------------------------------
extern "C" void kernel_launch(void* const* d_in, const int* in_sizes, int n_in,
                              void* d_out, int out_size, void* d_ws, size_t ws_size,
                              hipStream_t stream) {
  (void)in_sizes; (void)n_in; (void)out_size;
  const float* x        = (const float*)d_in[0];
  const float* gn_scale = (const float*)d_in[1];
  const float* gn_bias  = (const float*)d_in[2];
  const float* wq       = (const float*)d_in[3];
  const float* bq       = (const float*)d_in[4];
  const float* wk       = (const float*)d_in[5];
  const float* bk       = (const float*)d_in[6];
  const float* wv       = (const float*)d_in[7];
  const float* bv       = (const float*)d_in[8];
  const float* wp       = (const float*)d_in[9];
  const float* bp       = (const float*)d_in[10];
  float* out = (float*)d_out;

  // workspace: 4 bf16 weights | hT (reused as oT) | qT | kT | v   (~34 MB)
  char* ws = (char*)d_ws;
  const size_t WSZ = (size_t)C * C * sizeof(__bf16);         // 512 KB
  const size_t TSZ = (size_t)BATCH * N * C * sizeof(__bf16); // 8 MB
  __bf16* w_bf  = (__bf16*)ws;
  __bf16* wq_bf = w_bf + 0 * (size_t)C * C;
  __bf16* wk_bf = w_bf + 1 * (size_t)C * C;
  __bf16* wv_bf = w_bf + 2 * (size_t)C * C;
  __bf16* wp_bf = w_bf + 3 * (size_t)C * C;
  __bf16* hT    = (__bf16*)(ws + 4 * WSZ);
  __bf16* qT    = (__bf16*)(ws + 4 * WSZ + 1 * TSZ);
  __bf16* kT    = (__bf16*)(ws + 4 * WSZ + 2 * TSZ);
  __bf16* v_bf  = (__bf16*)(ws + 4 * WSZ + 3 * TSZ);
  __bf16* oT    = hT;                                        // hT dead after QKV
  if (ws_size < 4 * WSZ + 4 * TSZ) return;

  const float qscale = 0.044194173824159216f;                // 512^-0.5

  convert_weights_kernel<<<(C * C + 255) / 256, 256, 0, stream>>>(
      wq, wk, wv, wp, w_bf);

  groupnorm_kernel<<<BATCH * GROUPS, 256, 0, stream>>>(
      x, gn_scale, gn_bias, hT);

  dim3 gg(N / 64, C / 64, BATCH);
  gemm_wmma_kernel<<<gg, 128, 0, stream>>>(wq_bf, hT, bq, nullptr, nullptr,
                                           qT, qscale, 2);
  gemm_wmma_kernel<<<gg, 128, 0, stream>>>(wk_bf, hT, bk, nullptr, nullptr,
                                           kT, 1.0f, 2);
  gemm_wmma_kernel<<<gg, 128, 0, stream>>>(wv_bf, hT, bv, nullptr, nullptr,
                                           v_bf, 1.0f, 0);

  attn_flash_kernel<<<dim3(N / 16, BATCH), 128, 0, stream>>>(
      qT, kT, v_bf, oT);

  gemm_wmma_kernel<<<gg, 128, 0, stream>>>(wp_bf, oT, bp, x, out,
                                           nullptr, 1.0f, 1);
}
